// Network_1382979469731
// MI455X (gfx1250) — compile-verified, matrix operand order checked by reading, not run on hardware
//
#include <hip/hip_runtime.h>
#include <hip/hip_bf16.h>
#include <math.h>

// ---------------------------------------------------------------------------
// Problem constants (match reference)
// ---------------------------------------------------------------------------
#define N_NODES   50000
#define N_EDGES   600000
#define N_GRAPHS  100
#define MAX_R     3.5f

static constexpr float C_S      = 0.3826834323650898f;   // sin(pi/8)
static constexpr float C_X      = 0.9238795325112867f;   // cos(pi/8)
static constexpr float INV_NB   = 0.25f;                 // 1/sqrt(16)
static constexpr float RSQRT3   = 0.5773502691896258f;
static constexpr float SQRT3    = 1.7320508075688772f;
static constexpr float RS12     = 0.2886751345948129f;   // 1/sqrt(12)
static constexpr float RS100    = 0.1f;
static constexpr float RS16     = 0.25f;
static constexpr float RS32     = 0.1767766952966369f;
static constexpr float RS64     = 0.125f;
static constexpr float RS500    = 0.04472135954999579f;
static constexpr float SQ10     = 3.1622776601683795f;

typedef __attribute__((ext_vector_type(16))) _Float16 v16h;
typedef __attribute__((ext_vector_type(8)))  float    v8f;

// Fast transcendental helpers: v_exp_f32 + v_rcp_f32 (no IEEE div chains).
__device__ __forceinline__ float fast_rcp(float x) { return __builtin_amdgcn_rcpf(x); }
__device__ __forceinline__ float siluf(float x)    { return x * fast_rcp(1.f + __expf(-x)); }
__device__ __forceinline__ float sigmf(float x)    { return fast_rcp(1.f + __expf(-x)); }

// K-slot pattern of the 16-bit 16x32 A/B fragment (ISA 05_wmma §7.12.2):
// lanes 0-15 hold K {0..7,16..23}, lanes 16-31 hold K {8..15,24..31}
__device__ __forceinline__ int kpat(int lane, int j) {
    int b = (lane < 16) ? 0 : 8;
    return (j < 8) ? (b + j) : (16 + b + (j - 8));
}

__device__ __forceinline__ v16h load_frag16(const _Float16* p) {
    union { v16h v; uint4 u[2]; } r;
    const uint4* q = (const uint4*)p;
    r.u[0] = q[0];
    r.u[1] = q[1];
    return r.v;
}

// ---------------------------------------------------------------------------
// Pre-swizzle radial-MLP weights into WMMA-fragment-ready layout.
// W1f : [7 ntile][32 lane][16 slot]   halves  (K=12 padded->32, N=100 padded->112)
// W2f : [4 kk][8 ntile][32 lane][16]  halves  (K=100 padded->128, N=numel)
// Fan-in normalization (1/sqrt12, 1/sqrt100) baked in.
// ---------------------------------------------------------------------------
__global__ void prep_frags(const float* __restrict__ W1, const float* __restrict__ W2,
                           int numel, _Float16* __restrict__ W1f, _Float16* __restrict__ W2f)
{
    int t = blockIdx.x * blockDim.x + threadIdx.x;
    if (t < 7 * 32 * 16) {
        int j = t & 15, l = (t >> 4) & 31, nt = t >> 9;
        int n = nt * 16 + (l & 15);
        int k = kpat(l, j);
        float v = (k < 12 && n < 100) ? W1[k * 100 + n] * RS12 : 0.f;
        W1f[t] = (_Float16)v;
    }
    if (t < 4 * 8 * 32 * 16) {
        int j = t & 15, l = (t >> 4) & 31, nt = (t >> 9) & 7, kk = t >> 12;
        int n = nt * 16 + (l & 15);
        int k = kk * 32 + kpat(l, j);
        float v = (k < 100 && n < numel) ? W2[k * numel + n] * RS100 : 0.f;
        W2f[t] = (_Float16)v;
    }
}

// ---------------------------------------------------------------------------
// Per-edge geometry: r, cutoff*Y0 (e0), cutoff*Y1 (e1), radial features (f16,
// 12 real + padded to 16 halves so WMMA A-fragments load as one 16B vector).
// ---------------------------------------------------------------------------
__global__ void edge_geom(const float* __restrict__ evec, const float* __restrict__ z,
                          const int* __restrict__ esrc, const int* __restrict__ edst,
                          float* __restrict__ e0, float* __restrict__ e1,
                          _Float16* __restrict__ featH)
{
    int e = blockIdx.x * blockDim.x + threadIdx.x;
    if (e >= N_EDGES) return;
    float vx = evec[3*e+0], vy = evec[3*e+1], vz = evec[3*e+2];
    float r  = sqrtf(vx*vx + vy*vy + vz*vz);
    float rs = fmaxf(r, 1e-12f);
    // smooth cutoff
    float xx = r * (1.f / MAX_R);
    float u  = 2.f * (xx - 1.f);
    float y  = 0.5f * (1.f - __cosf(3.14159265358979f * u));
    float cut = (u > 0.f) ? 0.f : ((u < -1.f) ? 1.f : y);
    e0[e] = cut;
    float c = cut * SQRT3 * fast_rcp(rs);
    e1[3*e+0] = c * vx; e1[3*e+1] = c * vy; e1[3*e+2] = c * vz;
    // soft one-hot gaussian basis * sqrt(N_BASIS)
    const float step  = MAX_R / 9.f;
    const float rstep = 9.f / MAX_R;
    _Float16* f = featH + (long)e * 16;
    #pragma unroll
    for (int i = 0; i < 10; ++i) {
        float d = (r - (float)i * step) * rstep;
        f[i] = (_Float16)(__expf(-d * d) * (SQ10 / 1.12f));
    }
    f[10] = (_Float16)z[esrc[e]];
    f[11] = (_Float16)z[edst[e]];
    f[12] = (_Float16)0.f; f[13] = (_Float16)0.f; f[14] = (_Float16)0.f; f[15] = (_Float16)0.f;
}

// ---------------------------------------------------------------------------
// Fused edge kernel: WMMA radial MLP (16 edges per wave32) + message + scatter
// ---------------------------------------------------------------------------
#define WPB 4   // waves per block

template<int LAYER>
__global__ __launch_bounds__(128)
void edge_conv(const _Float16* __restrict__ featH,
               const _Float16* __restrict__ W1f, const _Float16* __restrict__ W2f,
               const float* __restrict__ e0,  const float* __restrict__ e1v,
               const float* __restrict__ s1,  const float* __restrict__ v1,
               const int*   __restrict__ esrc, const int* __restrict__ edst,
               float* __restrict__ aggS, float* __restrict__ aggV)
{
    constexpr int  M0    = (LAYER == 0) ? 16 : 32;             // scalar multiplicity
    constexpr int  NT2   = (LAYER == 0) ? 2 : ((LAYER == 1) ? 8 : 4); // numel/16
    constexpr bool SV    = (LAYER != 2);
    constexpr bool VS    = (LAYER == 1);
    constexpr bool VV    = (LAYER != 0);
    constexpr int  AGGSW = (LAYER == 0) ? 16 : 64;
    constexpr int  AGGVW = (LAYER == 0) ? 16 : 64;
    constexpr int  OFF_SS = 0;
    constexpr int  OFF_SV = M0;
    constexpr int  OFF_VS = 64;                                 // layer1 only
    constexpr int  OFF_VV = (LAYER == 1) ? 96 : 32;             // layer1 / layer2
    constexpr int  CPL   = M0 / 2;                              // channels per lane

    __shared__ _Float16 hS[WPB][16][128];   // GEMM1 out (silu'd), A for GEMM2
    __shared__ float    wS[WPB][16][128];   // per-edge TP weights

    const int lane = threadIdx.x & 31;
    const int wid  = threadIdx.x >> 5;
    const long tileBase = ((long)blockIdx.x * WPB + wid) * 16;
    if (tileBase >= N_EDGES) return;

    const int mrow = lane & 15;
    const int kh   = lane >> 4;
    const int base = kh * 8;

    long erow = tileBase + mrow;
    if (erow >= N_EDGES) erow = N_EDGES - 1;    // clamp; tail lanes masked later

    // ---- GEMM1: h[16x112] = silu(feat[16x12pad32] @ W1) --------------------
    union { v16h v; uint4 u[2]; } a1;
    a1.u[0] = *(const uint4*)(featH + erow * 16 + base);
    a1.u[1] = make_uint4(0u, 0u, 0u, 0u);

    #pragma unroll
    for (int nt = 0; nt < 7; ++nt) {
        v16h b = load_frag16(W1f + ((nt * 32) + lane) * 16);
        v8f  c = {};
        c = __builtin_amdgcn_wmma_f32_16x16x32_f16(false, a1.v, false, b,
                                                   (short)0, c, false, false);
        #pragma unroll
        for (int r = 0; r < 8; ++r)
            hS[wid][r + 8 * kh][nt * 16 + mrow] = (_Float16)siluf(c[r]);
    }
    #pragma unroll
    for (int j = 0; j < 8; ++j)                 // zero K-pad columns 112..127
        hS[wid][mrow][112 + base + j] = (_Float16)0.f;

    asm volatile("s_wait_dscnt 0" ::: "memory"); // wave-private LDS RAW

    // ---- A fragments for GEMM2 (K = hidden dim, 4 steps of 32) -------------
    union { v16h v; uint4 u[2]; } a2[4];
    #pragma unroll
    for (int kk = 0; kk < 4; ++kk) {
        a2[kk].u[0] = *(const uint4*)&hS[wid][mrow][kk * 32 + base];
        a2[kk].u[1] = *(const uint4*)&hS[wid][mrow][kk * 32 + 16 + base];
    }

    // ---- GEMM2: w[16 x numel] = h @ W2 -------------------------------------
    #pragma unroll
    for (int nt = 0; nt < NT2; ++nt) {
        v8f c = {};
        #pragma unroll
        for (int kk = 0; kk < 4; ++kk) {
            v16h b = load_frag16(W2f + (((kk * 8) + nt) * 32 + lane) * 16);
            c = __builtin_amdgcn_wmma_f32_16x16x32_f16(false, a2[kk].v, false, b,
                                                       (short)0, c, false, false);
        }
        #pragma unroll
        for (int r = 0; r < 8; ++r)
            wS[wid][r + 8 * kh][nt * 16 + mrow] = c[r];
    }
    asm volatile("s_wait_dscnt 0" ::: "memory");

    // ---- messages + scatter (lane = edge mrow, half kh of channels) --------
    const long e = tileBase + mrow;
    if (e < N_EDGES) {
        const int   src = esrc[e], dst = edst[e];
        const float ce0 = e0[e];
        const float ex  = e1v[3*e+0], ey = e1v[3*e+1], ez = e1v[3*e+2];
        const float* w  = wS[wid][mrow];
        #pragma unroll
        for (int t = 0; t < CPL; ++t) {
            const int cch = kh * CPL + t;
            const float ssrc = s1[(long)src * M0 + cch];
            // 0e x 0e -> 0e
            atomicAdd(&aggS[(long)dst * AGGSW + cch], w[OFF_SS + cch] * ssrc * ce0);
            if constexpr (SV) {   // 0e x 1o -> 1o
                float tv = w[OFF_SV + cch] * ssrc;
                float* p = &aggV[((long)dst * AGGVW + cch) * 3];
                atomicAdd(p + 0, tv * ex);
                atomicAdd(p + 1, tv * ey);
                atomicAdd(p + 2, tv * ez);
            }
            if constexpr (VS) {   // 1o x 0e -> 1o
                const float* vs = &v1[((long)src * 32 + cch) * 3];
                float wv = w[OFF_VS + cch] * ce0;
                float* p = &aggV[((long)dst * AGGVW + 32 + cch) * 3];
                atomicAdd(p + 0, wv * vs[0]);
                atomicAdd(p + 1, wv * vs[1]);
                atomicAdd(p + 2, wv * vs[2]);
            }
            if constexpr (VV) {   // 1o x 1o -> 0e
                const float* vs = &v1[((long)src * 32 + cch) * 3];
                float dot = vs[0] * ex + vs[1] * ey + vs[2] * ez;
                atomicAdd(&aggS[(long)dst * AGGSW + 32 + cch],
                          w[OFF_VV + cch] * dot * RSQRT3);
            }
        }
    }
}

// ---------------------------------------------------------------------------
// Node-side linear algebra (bandwidth-bound, VALU)
// ---------------------------------------------------------------------------
__global__ void lin_s_kernel(const float* __restrict__ in, int inW, int inStride,
                             const float* __restrict__ W, int outW,
                             float* __restrict__ out, int outStride, float scale)
{
    long t = (long)blockIdx.x * blockDim.x + threadIdx.x;
    if (t >= (long)N_NODES * outW) return;
    int o = (int)(t % outW); long n = t / outW;
    float acc = 0.f;
    for (int i = 0; i < inW; ++i) acc += in[n * inStride + i] * W[i * outW + o];
    out[n * outStride + o] = acc * scale;
}

__global__ void lin_v_kernel(const float* __restrict__ in, int inU,
                             const float* __restrict__ W, int outU,
                             float* __restrict__ out, float scale)
{
    long t = (long)blockIdx.x * blockDim.x + threadIdx.x;
    if (t >= (long)N_NODES * outU * 3) return;
    int d = (int)(t % 3); long t2 = t / 3;
    int o = (int)(t2 % outU); long n = t2 / outU;
    float acc = 0.f;
    for (int u = 0; u < inU; ++u) acc += in[(n * inU + u) * 3 + d] * W[u * outU + o];
    out[(n * outU + o) * 3 + d] = acc * scale;
}

__global__ void conv_out_s(const float* __restrict__ agg, int aggW,
                           const float* __restrict__ W, int outW,
                           const float* __restrict__ sc, float cs, float cx,
                           float* __restrict__ out, int outStride, float scale)
{
    long t = (long)blockIdx.x * blockDim.x + threadIdx.x;
    if (t >= (long)N_NODES * outW) return;
    int o = (int)(t % outW); long n = t / outW;
    float acc = 0.f;
    for (int c = 0; c < aggW; ++c) acc += agg[n * aggW + c] * W[c * outW + o];
    float skip = sc ? sc[n * outW + o] : 0.f;
    out[n * outStride + o] = cs * skip + cx * acc * scale;
}

__global__ void conv_out_v(const float* __restrict__ aggV, int aggW,
                           const float* __restrict__ W, int outW,
                           const float* __restrict__ scv, float cs, float cx,
                           float* __restrict__ outv, float scale)
{
    long t = (long)blockIdx.x * blockDim.x + threadIdx.x;
    if (t >= (long)N_NODES * outW * 3) return;
    int d = (int)(t % 3); long t2 = t / 3;
    int o = (int)(t2 % outW); long n = t2 / outW;
    float acc = 0.f;
    for (int c = 0; c < aggW; ++c) acc += aggV[(n * aggW + c) * 3 + d] * W[c * outW + o];
    float skip = scv ? scv[(n * outW + o) * 3 + d] : 0.f;
    outv[(n * outW + o) * 3 + d] = cs * skip + cx * acc * scale;
}

__global__ void gate_kernel(const float* __restrict__ convS,
                            float* __restrict__ sOut, float* __restrict__ v)
{
    long t = (long)blockIdx.x * blockDim.x + threadIdx.x;
    if (t >= (long)N_NODES * 32) return;
    int c = (int)(t % 32); long n = t / 32;
    sOut[n * 32 + c] = siluf(convS[n * 64 + c]);
    float g = sigmf(convS[n * 64 + 32 + c]);
    float* p = &v[(n * 32 + c) * 3];
    p[0] *= g; p[1] *= g; p[2] *= g;
}

__global__ void pool_kernel(const float* __restrict__ s, const int* __restrict__ batch,
                            float* __restrict__ out)
{
    int n = blockIdx.x * blockDim.x + threadIdx.x;
    if (n >= N_NODES) return;
    atomicAdd(&out[batch[n]], s[n] * RS500);
}

// ---------------------------------------------------------------------------
// Host orchestration
// ---------------------------------------------------------------------------
static inline int cdiv(long a, long b) { return (int)((a + b - 1) / b); }

extern "C" void kernel_launch(void* const* d_in, const int* in_sizes, int n_in,
                              void* d_out, int out_size, void* d_ws, size_t ws_size,
                              hipStream_t stream)
{
    // Input order: top-level dict insertion order; params pytree with sorted keys.
    const float* x        = (const float*)d_in[0];
    const float* z        = (const float*)d_in[1];
    const float* evec     = (const float*)d_in[2];
    const float* l0_fcw1  = (const float*)d_in[3];
    const float* l0_fcw2  = (const float*)d_in[4];
    const float* l0_lin1s = (const float*)d_in[5];
    const float* l0_lin2s = (const float*)d_in[6];
    const float* l0_lin2v = (const float*)d_in[7];
    const float* l0_scs   = (const float*)d_in[8];
    const float* l1_fcw1  = (const float*)d_in[9];
    const float* l1_fcw2  = (const float*)d_in[10];
    const float* l1_lin1s = (const float*)d_in[11];
    const float* l1_lin1v = (const float*)d_in[12];
    const float* l1_lin2s = (const float*)d_in[13];
    const float* l1_lin2v = (const float*)d_in[14];
    const float* l1_scs   = (const float*)d_in[15];
    const float* l1_scv   = (const float*)d_in[16];
    const float* l2_fcw1  = (const float*)d_in[17];
    const float* l2_fcw2  = (const float*)d_in[18];
    const float* l2_lin1s = (const float*)d_in[19];
    const float* l2_lin1v = (const float*)d_in[20];
    const float* l2_lin2s = (const float*)d_in[21];
    const float* l2_scs   = (const float*)d_in[22];
    const int*   esrc     = (const int*)d_in[23];
    const int*   edst     = (const int*)d_in[24];
    const int*   batch    = (const int*)d_in[25];

    // ---- workspace arena ----------------------------------------------------
    size_t off = 0;
    auto alloc = [&](size_t bytes) -> void* {
        void* p = (char*)d_ws + off;
        off = (off + bytes + 255) & ~(size_t)255;
        return p;
    };
    float*     e0    = (float*)alloc((size_t)N_EDGES * 4);
    float*     e1    = (float*)alloc((size_t)N_EDGES * 12);
    _Float16*  featH = (_Float16*)alloc((size_t)N_EDGES * 16 * 2);
    float*     s1    = (float*)alloc((size_t)N_NODES * 32 * 4);
    float*     v1    = (float*)alloc((size_t)N_NODES * 96 * 4);
    float*     scS   = (float*)alloc((size_t)N_NODES * 64 * 4);
    float*     scV   = (float*)alloc((size_t)N_NODES * 96 * 4);
    float*     aggS  = (float*)alloc((size_t)N_NODES * 64 * 4);
    float*     aggV  = (float*)alloc((size_t)N_NODES * 192 * 4);
    float*     convS = (float*)alloc((size_t)N_NODES * 64 * 4);
    float*     convV = (float*)alloc((size_t)N_NODES * 96 * 4);
    float*     sGate = (float*)alloc((size_t)N_NODES * 32 * 4);
    _Float16*  W1f   = (_Float16*)alloc(7 * 32 * 16 * 2);
    _Float16*  W2f   = (_Float16*)alloc(4 * 8 * 32 * 16 * 2);
    if (off > ws_size) return;   // workspace too small; bail deterministically

    const int TB = 256;
    const int edgeBlocks = cdiv(cdiv(N_EDGES, 16), WPB);

    hipMemsetAsync(d_out, 0, (size_t)out_size * 4, stream);

    // ---- edge geometry + radial features (shared across layers) ------------
    edge_geom<<<cdiv(N_EDGES, TB), TB, 0, stream>>>(evec, z, esrc, edst, e0, e1, featH);

    // ======================= LAYER 0 (paths ss, sv) ==========================
    lin_s_kernel<<<cdiv((long)N_NODES * 16, TB), TB, 0, stream>>>(x, 16, 16, l0_lin1s, 16, s1, 16, RS16);
    lin_s_kernel<<<cdiv((long)N_NODES * 64, TB), TB, 0, stream>>>(x, 16, 16, l0_scs, 64, scS, 64, RS16);
    prep_frags<<<64, TB, 0, stream>>>(l0_fcw1, l0_fcw2, 32, W1f, W2f);
    hipMemsetAsync(aggS, 0, (size_t)N_NODES * 64 * 4, stream);
    hipMemsetAsync(aggV, 0, (size_t)N_NODES * 192 * 4, stream);
    edge_conv<0><<<edgeBlocks, 128, 0, stream>>>(featH, W1f, W2f, e0, e1, s1, nullptr,
                                                 esrc, edst, aggS, aggV);
    conv_out_s<<<cdiv((long)N_NODES * 64, TB), TB, 0, stream>>>(aggS, 16, l0_lin2s, 64, scS, C_S, C_X, convS, 64, INV_NB * RS16);
    conv_out_v<<<cdiv((long)N_NODES * 96, TB), TB, 0, stream>>>(aggV, 16, l0_lin2v, 32, nullptr, 0.f, 1.f, convV, INV_NB * RS16);
    gate_kernel<<<cdiv((long)N_NODES * 32, TB), TB, 0, stream>>>(convS, sGate, convV);

    // ======================= LAYER 1 (ss, sv, vs, vv) ========================
    lin_s_kernel<<<cdiv((long)N_NODES * 32, TB), TB, 0, stream>>>(sGate, 32, 32, l1_lin1s, 32, s1, 32, RS32);
    lin_v_kernel<<<cdiv((long)N_NODES * 96, TB), TB, 0, stream>>>(convV, 32, l1_lin1v, 32, v1, RS32);
    lin_s_kernel<<<cdiv((long)N_NODES * 64, TB), TB, 0, stream>>>(sGate, 32, 32, l1_scs, 64, scS, 64, RS32);
    lin_v_kernel<<<cdiv((long)N_NODES * 96, TB), TB, 0, stream>>>(convV, 32, l1_scv, 32, scV, RS32);
    prep_frags<<<64, TB, 0, stream>>>(l1_fcw1, l1_fcw2, 128, W1f, W2f);
    hipMemsetAsync(aggS, 0, (size_t)N_NODES * 64 * 4, stream);
    hipMemsetAsync(aggV, 0, (size_t)N_NODES * 192 * 4, stream);
    edge_conv<1><<<edgeBlocks, 128, 0, stream>>>(featH, W1f, W2f, e0, e1, s1, v1,
                                                 esrc, edst, aggS, aggV);
    conv_out_s<<<cdiv((long)N_NODES * 64, TB), TB, 0, stream>>>(aggS, 64, l1_lin2s, 64, scS, C_S, C_X, convS, 64, INV_NB * RS64);
    conv_out_v<<<cdiv((long)N_NODES * 96, TB), TB, 0, stream>>>(aggV, 64, l1_lin2v, 32, scV, C_S, C_X, convV, INV_NB * RS64);
    gate_kernel<<<cdiv((long)N_NODES * 32, TB), TB, 0, stream>>>(convS, sGate, convV);

    // ======================= LAYER 2 (ss, vv) ================================
    lin_s_kernel<<<cdiv((long)N_NODES * 32, TB), TB, 0, stream>>>(sGate, 32, 32, l2_lin1s, 32, s1, 32, RS32);
    lin_v_kernel<<<cdiv((long)N_NODES * 96, TB), TB, 0, stream>>>(convV, 32, l2_lin1v, 32, v1, RS32);
    lin_s_kernel<<<cdiv((long)N_NODES * 1, TB), TB, 0, stream>>>(sGate, 32, 32, l2_scs, 1, scS, 1, RS32);
    prep_frags<<<64, TB, 0, stream>>>(l2_fcw1, l2_fcw2, 64, W1f, W2f);
    hipMemsetAsync(aggS, 0, (size_t)N_NODES * 64 * 4, stream);
    edge_conv<2><<<edgeBlocks, 128, 0, stream>>>(featH, W1f, W2f, e0, e1, s1, v1,
                                                 esrc, edst, aggS, nullptr);
    conv_out_s<<<cdiv((long)N_NODES * 1, TB), TB, 0, stream>>>(aggS, 64, l2_lin2s, 1, scS, C_S, C_X, convS, 1, INV_NB * RS64);

    // ======================= graph pooling ===================================
    pool_kernel<<<cdiv(N_NODES, TB), TB, 0, stream>>>(convS, batch, (float*)d_out);
}
